// MultiHeadLatentAttention_45578192945288
// MI455X (gfx1250) — matrix operand order repaired and checked by reading, not compiled, not run on hardware
//
#include <hip/hip_runtime.h>
#include <hip/hip_bf16.h>
#include <math.h>

// ---------------- problem constants ----------------
#define B_   2
#define T_   2048
#define DM_  2048
#define H_   16
#define HD_  128
#define KVL_ 512
#define QL_  1024
#define R_   64
#define E_   192           // HD_ + R_
#define BT_  (B_ * T_)     // 4096

typedef __bf16 v16bf __attribute__((ext_vector_type(16)));
typedef float  v8f   __attribute__((ext_vector_type(8)));
typedef unsigned v4u __attribute__((ext_vector_type(4)));
typedef int      v4i __attribute__((ext_vector_type(4)));
typedef int      v8i __attribute__((ext_vector_type(8)));

union Frag { unsigned u[8]; v16bf v; };

__device__ __forceinline__ unsigned short f2bf(float f) {
  unsigned u = __builtin_bit_cast(unsigned, f);
  unsigned r = u + 0x7FFFu + ((u >> 16) & 1u);   // round-to-nearest-even
  return (unsigned short)(r >> 16);
}

// ---- CDNA5 async global->LDS (ASYNCcnt path) ----
__device__ __forceinline__ void async_load_b128(unsigned lds_off, const void* gaddr) {
  asm volatile("global_load_async_to_lds_b128 %0, %1, off"
               :: "v"(lds_off), "v"(gaddr) : "memory");
}
__device__ __forceinline__ void wait_async0() {
  asm volatile("s_wait_asynccnt 0x0" ::: "memory");
}

// ---- CDNA5 Tensor Data Mover: 2-D tile (32 cols x 128 rows, bf16) -> LDS ----
// D# group0: count=1, lds_addr, global tile addr, type=2.
// D# group1: data_size=2B, tensor_dim0=K, tile_dim0=32, tile_dim1=128,
//            tensor_dim0_stride=K.
__device__ __forceinline__ void tdm_load_tile_2d(unsigned lds_off,
                                                 unsigned long long gaddr,
                                                 unsigned K_elems) {
  v4u g0;
  g0.x = 1u;                                            // count=1, user mode
  g0.y = lds_off;                                       // lds_addr [63:32]
  g0.z = (unsigned)(gaddr & 0xFFFFFFFFull);             // global_addr lo
  g0.w = (unsigned)((gaddr >> 32) & 0x01FFFFFFull) | (2u << 30);  // type=2
  const unsigned td1 = 1u << 20;                        // tensor_dim1 (OOB only)
  v8i g1;
  g1[0] = 0x00010000;                                   // data_size=1 (2 bytes)
  g1[1] = (int)((K_elems & 0xFFFFu) << 16);             // tensor_dim0 lo -> [79:48]
  g1[2] = (int)((K_elems >> 16) | ((td1 & 0xFFFFu) << 16));
  g1[3] = (int)((td1 >> 16) | (32u << 16));             // tile_dim0 = 32
  g1[4] = 128;                                          // tile_dim1 = 128
  g1[5] = (int)K_elems;                                 // tensor_dim0_stride lo
  g1[6] = 0;
  g1[7] = 0;
#if __clang_major__ >= 23
  __builtin_amdgcn_tensor_load_to_lds(g0, g1, (v4i)0, (v4i)0, (v8i)0, 0);
#else
  __builtin_amdgcn_tensor_load_to_lds(g0, g1, (v4i)0, (v4i)0, 0);
#endif
}

// ---------------- fp32 -> bf16 convert ----------------
__global__ void f2b_kernel(const float* __restrict__ in,
                           unsigned short* __restrict__ out, int n) {
  int i = blockIdx.x * blockDim.x + threadIdx.x;
  if (i < n) out[i] = f2bf(in[i]);
}

// ---------------- generic bf16 WMMA GEMM ----------------
// C[M,N] = A[M,K] @ B[K,N]; A,B row-major bf16. Block tile 128x64, 8 waves,
// each wave: 16 rows x 64 cols (4 wmma accumulators). A tile staged by TDM.
// Cf (optional): plain row-major f32, ld = N.
// Cb (optional): bf16; mapMode 0: plain row-major ld = ldo.
//                mapMode 1: head-major scatter (hd fixed at 128).
__global__ __launch_bounds__(256)
void gemm_bf16(const unsigned short* __restrict__ A,
               const unsigned short* __restrict__ Bm,
               float* __restrict__ Cf,
               unsigned short* __restrict__ Cb,
               int M, int N, int K,
               int mapMode, int ldo, int colOff) {
  __shared__ unsigned Alds[128 * 16];   // 128 rows x 16 u32 (=32 bf16 along K)
  __shared__ unsigned Blds[16 * 64];    // 16 K-pairs x 64 cols, pair-packed u32

  const int tid  = threadIdx.x;
  const int lane = tid & 31;
  const int wid  = tid >> 5;       // 0..7 -> 16-row strip
  const int half = lane >> 4;
  const int l16  = lane & 15;

  const int mTile = blockIdx.y * 128;
  const int nTile = blockIdx.x * 64;

  v8f acc[4];
#pragma unroll
  for (int i = 0; i < 4; ++i) acc[i] = (v8f){};

  const unsigned ldsA0 = (unsigned)(size_t)&Alds[0];

  for (int k0 = 0; k0 < K; k0 += 32) {
    __syncthreads();
    if (wid == 0) {                      // one wave drives the TDM
      unsigned long long ga =
          (unsigned long long)(size_t)(A + (size_t)mTile * K + k0);
      tdm_load_tile_2d(ldsA0, ga, (unsigned)K);
      __builtin_amdgcn_s_wait_tensorcnt((short)0);
    }
#pragma unroll
    for (int i = 0; i < 4; ++i) {        // B tile: pack (2k,2k+1) pairs
      int e = tid + i * 256;
      int kp = e >> 6, n = e & 63;
      unsigned short b0 = Bm[(size_t)(k0 + 2 * kp) * N + nTile + n];
      unsigned short b1 = Bm[(size_t)(k0 + 2 * kp + 1) * N + nTile + n];
      Blds[e] = (unsigned)b0 | ((unsigned)b1 << 16);
    }
    __syncthreads();

    Frag a;
#pragma unroll
    for (int v = 0; v < 8; ++v) {
      int k2 = v + 8 * half;   // A frag layout
      a.u[v] = Alds[(wid * 16 + l16) * 16 + k2];
    }
#pragma unroll
    for (int ns = 0; ns < 4; ++ns) {
      Frag bf;
#pragma unroll
      for (int v = 0; v < 8; ++v)
        bf.u[v] = Blds[(v + 8 * half) * 64 + ns * 16 + l16];
      acc[ns] = __builtin_amdgcn_wmma_f32_16x16x32_bf16(
          false, a.v, false, bf.v, (short)0, acc[ns], false, false);
    }
  }

#pragma unroll
  for (int ns = 0; ns < 4; ++ns)
#pragma unroll
    for (int r = 0; r < 8; ++r) {        // C layout: M=r+8*half, N=l16
      int row = mTile + wid * 16 + r + 8 * half;
      int col = nTile + ns * 16 + l16;
      float f = acc[ns][r];
      if (Cf) Cf[(size_t)row * N + col] = f;
      if (Cb) {
        if (mapMode == 0) {
          Cb[(size_t)row * ldo + col] = f2bf(f);
        } else {                         // hd == 128: shifts, no division
          int bI = row >> 11, t = row & (T_ - 1);
          int h = col >> 7, d = col & (HD_ - 1);
          Cb[((size_t)(bI * H_ + h) * T_ + t) * ldo + colOff + d] = f2bf(f);
        }
      }
    }
}

// ---------------- rope kernels ----------------
__global__ void rope_kr_kernel(const float* __restrict__ pre,
                               const float* __restrict__ freqs,
                               float* __restrict__ outKr,
                               unsigned short* __restrict__ Kbuf) {
  int i = blockIdx.x * blockDim.x + threadIdx.x;
  if (i >= BT_ * (R_ / 2)) return;
  int bt = i >> 5, j = i & 31;
  int b = bt / T_, t = bt % T_;
  float ang = freqs[t * (R_ / 2) + j];
  float c = cosf(ang), s = sinf(ang);
  float x0 = pre[(size_t)bt * R_ + 2 * j];
  float x1 = pre[(size_t)bt * R_ + 2 * j + 1];
  float y0 = x0 * c - x1 * s;
  float y1 = x0 * s + x1 * c;
  outKr[(size_t)bt * R_ + 2 * j]     = y0;
  outKr[(size_t)bt * R_ + 2 * j + 1] = y1;
  unsigned short b0 = f2bf(y0), b1 = f2bf(y1);
#pragma unroll
  for (int h = 0; h < H_; ++h) {             // broadcast shared rope key
    size_t base = ((size_t)(b * H_ + h) * T_ + t) * E_ + HD_ + 2 * j;
    Kbuf[base] = b0; Kbuf[base + 1] = b1;
  }
}

__global__ void rope_qr_kernel(const float* __restrict__ pre,
                               const float* __restrict__ freqs,
                               unsigned short* __restrict__ Qbuf) {
  int i = blockIdx.x * blockDim.x + threadIdx.x;
  if (i >= BT_ * H_ * (R_ / 2)) return;
  int bt = i / (H_ * (R_ / 2));
  int rem = i % (H_ * (R_ / 2));
  int h = rem >> 5, j = rem & 31;
  int b = bt / T_, t = bt % T_;
  float ang = freqs[t * (R_ / 2) + j];
  float c = cosf(ang), s = sinf(ang);
  float x0 = pre[(size_t)bt * (H_ * R_) + h * R_ + 2 * j];
  float x1 = pre[(size_t)bt * (H_ * R_) + h * R_ + 2 * j + 1];
  size_t base = ((size_t)(b * H_ + h) * T_ + t) * E_ + HD_ + 2 * j;
  Qbuf[base]     = f2bf(x0 * c - x1 * s);
  Qbuf[base + 1] = f2bf(x0 * s + x1 * c);
}

// ---------------- flash attention (bf16 WMMA, causal) ----------------
// Q,K: [b*H+h][t][192] bf16; V: [b*H+h][t][128] bf16.
// Output: bf16 plain [b*T+t][h*128+d] (feeds the W_O GEMM).
__global__ __launch_bounds__(256)
void mla_attn_kernel(const unsigned short* __restrict__ Q,
                     const unsigned short* __restrict__ Kb,
                     const unsigned short* __restrict__ Vb,
                     unsigned short* __restrict__ Ob) {
  __shared__ unsigned Klds[32 * 96];     // 32 keys x 96 u32 (192 bf16), 12KB
  __shared__ unsigned Vlds[16 * 128];    // key-pairs packed u32 x 128 cols
  __shared__ unsigned Plds[8][16 * 16];  // per-wave P: 16 x 32 bf16

  const int tid  = threadIdx.x;
  const int lane = tid & 31;
  const int wid  = tid >> 5;
  const int half = lane >> 4;
  const int l16  = lane & 15;
  const int bh   = blockIdx.y;                    // 0..B*H-1
  const size_t tBase = (size_t)bh * T_;
  const int qBlock = blockIdx.x * 128;
  const int qWave  = qBlock + wid * 16;           // this wave's 16 query rows

  // load Q A-fragments: 6 chunks of K=32 along E=192
  Frag qf[6];
  {
    const unsigned* Qg = (const unsigned*)Q;
    size_t rowBase = (tBase + qWave + l16) * (E_ / 2);
#pragma unroll
    for (int c = 0; c < 6; ++c)
#pragma unroll
      for (int v = 0; v < 8; ++v) {
        int k2 = c * 16 + v + 8 * half;
        qf[c].u[v] = Qg[rowBase + k2];
      }
  }

  v8f o[8];
#pragma unroll
  for (int f = 0; f < 8; ++f) o[f] = (v8f){};
  float mrow[8], lrow[8];
#pragma unroll
  for (int r = 0; r < 8; ++r) { mrow[r] = -3e38f; lrow[r] = 0.f; }

  const float scale = 0.07216878364870323f;       // 1/sqrt(192)
  const int kEnd = qBlock + 128;
  const unsigned kldsBase = (unsigned)(size_t)&Klds[0];

  for (int k0 = 0; k0 < kEnd; k0 += 32) {
    __syncthreads();
    // K tile is a contiguous 12KB span: async b128 copies, 3 per thread
    const char* kg = (const char*)Kb + (tBase + k0) * (E_ * 2);
#pragma unroll
    for (int i = 0; i < 3; ++i) {
      unsigned off = (unsigned)(tid + i * 256) * 16u;
      async_load_b128(kldsBase + off, kg + off);
    }
#pragma unroll
    for (int i = 0; i < 8; ++i) {                 // V tile: pair-pack
      int e = tid + i * 256;
      int p = e >> 7, n = e & 127;
      unsigned short v0 = Vb[(tBase + k0 + 2 * p) * HD_ + n];
      unsigned short v1 = Vb[(tBase + k0 + 2 * p + 1) * HD_ + n];
      Vlds[e] = (unsigned)v0 | ((unsigned)v1 << 16);
    }
    wait_async0();
    __syncthreads();

    // S = Q K^T : two 16x16 tiles (keys k0..k0+15, k0+16..k0+31)
    v8f s0 = (v8f){}, s1 = (v8f){};
#pragma unroll
    for (int c = 0; c < 6; ++c) {
      Frag b0, b1;
#pragma unroll
      for (int v = 0; v < 8; ++v) {
        int k2 = c * 16 + v + 8 * half;           // B frag: pairs along E
        b0.u[v] = Klds[l16 * 96 + k2];
        b1.u[v] = Klds[(16 + l16) * 96 + k2];
      }
      s0 = __builtin_amdgcn_wmma_f32_16x16x32_bf16(false, qf[c].v, false, b0.v,
                                                   (short)0, s0, false, false);
      s1 = __builtin_amdgcn_wmma_f32_16x16x32_bf16(false, qf[c].v, false, b1.v,
                                                   (short)0, s1, false, false);
    }

    // causal mask + online softmax (row = r + 8*half lives on 16-lane half)
    float alpha[8];
    unsigned short* Pw = (unsigned short*)Plds[wid];
#pragma unroll
    for (int r = 0; r < 8; ++r) {
      int q = qWave + r + 8 * half;
      float v0 = (k0 + l16      <= q) ? s0[r] * scale : -3e38f;
      float v1 = (k0 + 16 + l16 <= q) ? s1[r] * scale : -3e38f;
      float mx = fmaxf(v0, v1);
#pragma unroll
      for (int d = 8; d >= 1; d >>= 1) mx = fmaxf(mx, __shfl_xor(mx, d, 16));
      float mnew = fmaxf(mrow[r], mx);
      float p0 = __expf(v0 - mnew);
      float p1 = __expf(v1 - mnew);
      float rs = p0 + p1;
#pragma unroll
      for (int d = 8; d >= 1; d >>= 1) rs += __shfl_xor(rs, d, 16);
      alpha[r] = __expf(mrow[r] - mnew);
      lrow[r] = lrow[r] * alpha[r] + rs;
      mrow[r] = mnew;
      int m = r + 8 * half;
      Pw[m * 32 + l16]      = f2bf(p0);
      Pw[m * 32 + 16 + l16] = f2bf(p1);
    }
#pragma unroll
    for (int f = 0; f < 8; ++f)
#pragma unroll
      for (int r = 0; r < 8; ++r) o[f][r] *= alpha[r];

    __syncthreads();

    // O += P @ V
    Frag pf;
#pragma unroll
    for (int v = 0; v < 8; ++v) {
      int k2 = v + 8 * half;
      pf.u[v] = Plds[wid][l16 * 16 + k2];
    }
#pragma unroll
    for (int f = 0; f < 8; ++f) {
      Frag vf;
#pragma unroll
      for (int v = 0; v < 8; ++v)
        vf.u[v] = Vlds[(v + 8 * half) * 128 + f * 16 + l16];
      o[f] = __builtin_amdgcn_wmma_f32_16x16x32_bf16(false, pf.v, false, vf.v,
                                                     (short)0, o[f], false, false);
    }
  }

  // normalize and store (plain [b*T+t][h*128+d] bf16)
  const int b = bh >> 4, h = bh & 15;
#pragma unroll
  for (int f = 0; f < 8; ++f)
#pragma unroll
    for (int r = 0; r < 8; ++r) {
      int m = r + 8 * half;
      size_t row = (size_t)b * T_ + (qWave + m);
      Ob[row * (H_ * HD_) + h * HD_ + f * 16 + l16] = f2bf(o[f][r] / lrow[r]);
    }
}

// ---------------- host launcher ----------------
extern "C" void kernel_launch(void* const* d_in, const int* in_sizes, int n_in,
                              void* d_out, int out_size, void* d_ws, size_t ws_size,
                              hipStream_t stream) {
  (void)in_sizes; (void)n_in; (void)out_size; (void)ws_size;
  const float* x      = (const float*)d_in[0];
  const float* freqs  = (const float*)d_in[1];
  const float* W_DKV  = (const float*)d_in[2];
  const float* W_UK   = (const float*)d_in[3];
  const float* W_UV   = (const float*)d_in[4];
  const float* W_KR   = (const float*)d_in[5];
  const float* W_DQ   = (const float*)d_in[6];
  const float* W_UQ   = (const float*)d_in[7];
  const float* W_QR   = (const float*)d_in[8];
  const float* W_O    = (const float*)d_in[9];

  float* out_y   = (float*)d_out;                          // B*T*DM
  float* out_ckv = out_y + (size_t)BT_ * DM_;              // B*T*KVL
  float* out_kr  = out_ckv + (size_t)BT_ * KVL_;           // B*T*R

  char* ws = (char*)d_ws;
  size_t off = 0;
  auto alloc = [&](size_t bytes) -> char* {
    char* p = ws + off;
    off += (bytes + 255) & ~(size_t)255;
    return p;
  };
  unsigned short* xb    = (unsigned short*)alloc((size_t)BT_ * DM_ * 2);
  unsigned short* wdkv  = (unsigned short*)alloc((size_t)DM_ * KVL_ * 2);
  unsigned short* wuk   = (unsigned short*)alloc((size_t)KVL_ * H_ * HD_ * 2);
  unsigned short* wuv   = (unsigned short*)alloc((size_t)KVL_ * H_ * HD_ * 2);
  unsigned short* wkr   = (unsigned short*)alloc((size_t)DM_ * R_ * 2);
  unsigned short* wdq   = (unsigned short*)alloc((size_t)DM_ * QL_ * 2);
  unsigned short* wuq   = (unsigned short*)alloc((size_t)QL_ * H_ * HD_ * 2);
  unsigned short* wqr   = (unsigned short*)alloc((size_t)QL_ * H_ * R_ * 2);
  unsigned short* wo    = (unsigned short*)alloc((size_t)H_ * HD_ * DM_ * 2);
  unsigned short* ckvb  = (unsigned short*)alloc((size_t)BT_ * KVL_ * 2);
  unsigned short* cqb   = (unsigned short*)alloc((size_t)BT_ * QL_ * 2);
  float*          krpre = (float*)alloc((size_t)BT_ * R_ * 4);
  float*          qrpre = (float*)alloc((size_t)BT_ * H_ * R_ * 4);
  unsigned short* Kbuf  = (unsigned short*)alloc((size_t)B_ * H_ * T_ * E_ * 2);
  unsigned short* Qbuf  = (unsigned short*)alloc((size_t)B_ * H_ * T_ * E_ * 2);
  unsigned short* Vbuf  = (unsigned short*)alloc((size_t)B_ * H_ * T_ * HD_ * 2);
  unsigned short* aob   = (unsigned short*)alloc((size_t)BT_ * H_ * HD_ * 2);

  auto cvt = [&](const float* src, unsigned short* dst, size_t n) {
    f2b_kernel<<<dim3((unsigned)((n + 255) / 256)), dim3(256), 0, stream>>>(src, dst, (int)n);
  };
  cvt(x,     xb,   (size_t)BT_ * DM_);
  cvt(W_DKV, wdkv, (size_t)DM_ * KVL_);
  cvt(W_UK,  wuk,  (size_t)KVL_ * H_ * HD_);
  cvt(W_UV,  wuv,  (size_t)KVL_ * H_ * HD_);
  cvt(W_KR,  wkr,  (size_t)DM_ * R_);
  cvt(W_DQ,  wdq,  (size_t)DM_ * QL_);
  cvt(W_UQ,  wuq,  (size_t)QL_ * H_ * HD_);
  cvt(W_QR,  wqr,  (size_t)QL_ * H_ * R_);
  cvt(W_O,   wo,   (size_t)H_ * HD_ * DM_);

  auto gemm = [&](const unsigned short* A, const unsigned short* Bm,
                  float* Cf, unsigned short* Cb, int M, int N, int K,
                  int mapMode, int ldo, int colOff) {
    gemm_bf16<<<dim3(N / 64, M / 128), dim3(256), 0, stream>>>(
        A, Bm, Cf, Cb, M, N, K, mapMode, ldo, colOff);
  };

  // c_KV = x @ W_DKV  (f32 -> d_out, bf16 -> ckvb)
  gemm(xb, wdkv, out_ckv, ckvb, BT_, KVL_, DM_, 0, KVL_, 0);
  // K_r pre-rope = x @ W_KR
  gemm(xb, wkr, krpre, nullptr, BT_, R_, DM_, 0, R_, 0);
  // c_Q = x @ W_DQ
  gemm(xb, wdq, nullptr, cqb, BT_, QL_, DM_, 0, QL_, 0);
  // K_c = c_KV @ W_UK -> head-major K buffer cols [0,128)
  gemm(ckvb, wuk, nullptr, Kbuf, BT_, H_ * HD_, KVL_, 1, E_, 0);
  // V = c_KV @ W_UV -> head-major V buffer
  gemm(ckvb, wuv, nullptr, Vbuf, BT_, H_ * HD_, KVL_, 1, HD_, 0);
  // Q_c = c_Q @ W_UQ -> head-major Q buffer cols [0,128)
  gemm(cqb, wuq, nullptr, Qbuf, BT_, H_ * HD_, QL_, 1, E_, 0);
  // Q_r pre-rope = c_Q @ W_QR
  gemm(cqb, wqr, qrpre, nullptr, BT_, H_ * R_, QL_, 0, H_ * R_, 0);

  // rope: K_r (f32 out + bf16 broadcast into Kbuf cols [128,192))
  rope_kr_kernel<<<dim3((BT_ * (R_ / 2) + 255) / 256), dim3(256), 0, stream>>>(
      krpre, freqs, out_kr, Kbuf);
  // rope: Q_r -> Qbuf cols [128,192)
  rope_qr_kernel<<<dim3((BT_ * H_ * (R_ / 2) + 255) / 256), dim3(256), 0, stream>>>(
      qrpre, freqs, Qbuf);

  // causal flash attention
  mla_attn_kernel<<<dim3(T_ / 128, B_ * H_), dim3(256), 0, stream>>>(
      Qbuf, Kbuf, Vbuf, aob);

  // y = attn_out @ W_O
  gemm(aob, wo, out_y, nullptr, BT_, DM_, H_ * HD_, 0, DM_, 0);
}
